// S4DBlock_4853313044574
// MI455X (gfx1250) — matrix operand order repaired and checked by reading, not compiled, not run on hardware
//
#include <hip/hip_runtime.h>
#include <math.h>

// ---------------------------------------------------------------- types
typedef __bf16  bf16_t;
typedef __attribute__((ext_vector_type(16))) __bf16 v16bf;
typedef __attribute__((ext_vector_type(8)))  float  v8f;

struct Q2 { uint4 lo; uint4 hi; };          // 32 bytes == v16bf
union FragU { Q2 q; v16bf v; };

static __device__ __forceinline__ v16bf make_frag(const unsigned short* p0,
                                                  const unsigned short* p1) {
    FragU u;
    u.q.lo = *reinterpret_cast<const uint4*>(p0);
    u.q.hi = *reinterpret_cast<const uint4*>(p1);
    return u.v;
}

static __device__ __forceinline__ unsigned short f2bf(float f) {
    unsigned int u = __float_as_uint(f);
    u = (u + 0x7FFFu + ((u >> 16) & 1u)) >> 16;   // round-to-nearest-even
    return (unsigned short)u;
}

static __device__ __forceinline__ float softplus_f(float x) {
    return (x > 20.f) ? x : log1pf(expf(x));
}

static __device__ __forceinline__ float gelu_exact(float x) {
    return 0.5f * x * (1.0f + erff(x * 0.70710678118654752f));
}

// 256-thread (8 wave32) block sum; buf must hold >= 8 floats in LDS
static __device__ __forceinline__ float block_sum(float v, float* buf) {
    for (int o = 16; o > 0; o >>= 1) v += __shfl_down(v, o);
    int lane = threadIdx.x & 31, wid = threadIdx.x >> 5;
    if (lane == 0) buf[wid] = v;
    __syncthreads();
    float r = 0.f;
    if (threadIdx.x < 8) r = buf[threadIdx.x];
    if (wid == 0) {
        for (int o = 4; o > 0; o >>= 1) r += __shfl_down(r, o);
        if (lane == 0) buf[0] = r;
    }
    __syncthreads();
    r = buf[0];
    __syncthreads();
    return r;
}

// ---------------------------------------------------------------- problem dims
#define BB 4
#define TT 4096
#define CC 1024
#define NN 64
#define HH 4096
#define MM (BB * TT)          // 16384 tokens

// ---------------------------------------------------------------- kernels

// f32 -> bf16 weight conversion
__global__ __launch_bounds__(256)
void k_cvt_bf16(const float* __restrict__ in, unsigned short* __restrict__ out, int n) {
    int i = blockIdx.x * 256 + threadIdx.x;
    int stride = gridDim.x * 256;
    for (; i < n; i += stride) out[i] = f2bf(in[i]);
}

// LayerNorm over C=1024; one block per token; writes f32 (row-major) or bf16
__global__ __launch_bounds__(256)
void k_layernorm(const float* __restrict__ in, const float* __restrict__ w,
                 const float* __restrict__ b, float* __restrict__ outf,
                 unsigned short* __restrict__ outbf) {
    __shared__ float red[8];
    const size_t tok = blockIdx.x;
    const float* row = in + tok * CC;
    float x4[4];
    float s = 0.f;
#pragma unroll
    for (int i = 0; i < 4; ++i) {
        x4[i] = row[threadIdx.x + i * 256];
        s += x4[i];
    }
    const float mu = block_sum(s, red) * (1.0f / CC);
    float sq = 0.f;
#pragma unroll
    for (int i = 0; i < 4; ++i) {
        float d = x4[i] - mu;
        sq += d * d;
    }
    const float rstd = rsqrtf(block_sum(sq, red) * (1.0f / CC) + 1e-5f);
#pragma unroll
    for (int i = 0; i < 4; ++i) {
        int c = threadIdx.x + i * 256;
        float v = (x4[i] - mu) * rstd * w[c] + b[c];
        if (outf)  outf[tok * CC + c] = v;
        else       outbf[tok * CC + c] = f2bf(v);
    }
}

// 32x32 LDS-tiled transpose (+ optional elementwise add):
// dst[b][q][p] = src[b][p][q] + (add ? add[b][q][p] : 0)
__global__ __launch_bounds__(256)
void k_transpose_add(const float* __restrict__ src, const float* __restrict__ add,
                     float* __restrict__ dst, int P, int Q) {
    __shared__ float tile[32][33];
    const int b  = blockIdx.z;
    const int q0 = blockIdx.x * 32;
    const int p0 = blockIdx.y * 32;
    const size_t planeS = (size_t)P * Q;
    const float* sp = src + (size_t)b * planeS;
    float*       dp = dst + (size_t)b * planeS;
    const int tx = threadIdx.x, ty = threadIdx.y;
#pragma unroll
    for (int i = 0; i < 4; ++i)
        tile[ty + i * 8][tx] = sp[(size_t)(p0 + ty + i * 8) * Q + q0 + tx];
    __syncthreads();
#pragma unroll
    for (int i = 0; i < 4; ++i) {
        const size_t idx = (size_t)(q0 + ty + i * 8) * P + p0 + tx;
        float v = tile[tx][ty + i * 8];
        if (add) v += add[(size_t)b * planeS + idx];
        dp[idx] = v;
    }
}

// S4D diagonal-SSM scan. h is (B,C,T) f32, processed IN PLACE:
//   out[t] = sum_j mix[c,j]*s_j[t] + D[c]*u[t],  s_j[t] = a_j*s_j[t-1] + u[t]
// One wave32 per (b,c); each lane owns modes {lane, lane+32}.
__global__ __launch_bounds__(32)
void k_s4d_scan(float* __restrict__ h, const float* __restrict__ log_tau,
                const float* __restrict__ log_dt, const float* __restrict__ mix,
                const float* __restrict__ D) {
    const int lane = threadIdx.x;
    const int bc   = blockIdx.x;           // b*C + c
    const int c    = bc & (CC - 1);
    const float dt = softplus_f(log_dt[0]);
    const float a0 = expf(-softplus_f(log_tau[lane])      * dt);
    const float a1 = expf(-softplus_f(log_tau[lane + 32]) * dt);
    const float w0 = mix[c * NN + lane];
    const float w1 = mix[c * NN + lane + 32];
    const float Dc = D[c];
    float* row = h + (size_t)bc * TT;
    float s0 = 0.f, s1 = 0.f;
    for (int t0 = 0; t0 < TT; t0 += 32) {
        const float u = row[t0 + lane];    // 32 time steps in registers
        float yv = 0.f;
#pragma unroll
        for (int k = 0; k < 32; ++k) {
            const float uk = __shfl(u, k);
            s0 = fmaf(a0, s0, uk);
            s1 = fmaf(a1, s1, uk);
            float p = fmaf(w1, s1, w0 * s0);
            p += __shfl_xor(p, 16);
            p += __shfl_xor(p, 8);
            p += __shfl_xor(p, 4);
            p += __shfl_xor(p, 2);
            p += __shfl_xor(p, 1);
            if (k == lane) yv = p;         // lane k keeps y[t0+k]
        }
        row[t0 + lane] = fmaf(Dc, u, yv);  // in place (u already consumed)
    }
}

// bf16 WMMA GEMM:  out[M,N] = epilogue( A[M,K] * W[N,K]^T + bias )
// Block 256 thr = 8 waves, 128x128 tile; wave tile 32(M) x 64(N); K step 32.
template <bool GELU_OUT>
__global__ __launch_bounds__(256)
void k_gemm_wmma(const unsigned short* __restrict__ A,   // bf16 M x K
                 const unsigned short* __restrict__ W,   // bf16 N x K
                 const float* __restrict__ bias,         // N
                 const float* __restrict__ resid,        // M x N (GELU_OUT=false)
                 unsigned short* __restrict__ outb,      // bf16 out (GELU_OUT=true)
                 float* __restrict__ outf,               // f32 out  (GELU_OUT=false)
                 int K, int N) {
    __shared__ __align__(16) unsigned short lA[128 * 40];   // 32-wide + 8-half pad
    __shared__ __align__(16) unsigned short lB[128 * 40];

    const int tid  = threadIdx.x;
    const int lane = tid & 31;
    const int wid  = tid >> 5;
    const int wm   = wid >> 1;             // 0..3  -> M offset wm*32
    const int wn   = wid & 1;              // 0..1  -> N offset wn*64
    const int hi   = lane >> 4;            // lane half
    const int l16  = lane & 15;

    const int bm = blockIdx.y * 128;
    const int bn = blockIdx.x * 128;

    const v8f zero = {0.f, 0.f, 0.f, 0.f, 0.f, 0.f, 0.f, 0.f};
    v8f acc[2][4];
#pragma unroll
    for (int mt = 0; mt < 2; ++mt)
#pragma unroll
        for (int nt = 0; nt < 4; ++nt) acc[mt][nt] = zero;

    const int row  = tid >> 1;             // 0..127
    const int koff = (tid & 1) << 4;       // 0 or 16 halfs

    for (int k0 = 0; k0 < K; k0 += 32) {
        const unsigned short* ga = A + (size_t)(bm + row) * K + k0 + koff;
        const unsigned short* gw = W + (size_t)(bn + row) * K + k0 + koff;
        uint4 a0 = *reinterpret_cast<const uint4*>(ga);
        uint4 a1 = *reinterpret_cast<const uint4*>(ga + 8);
        uint4 b0 = *reinterpret_cast<const uint4*>(gw);
        uint4 b1 = *reinterpret_cast<const uint4*>(gw + 8);
        if (k0 + 32 < K) {                 // gfx1250 global_prefetch_b8 of next K tile
            __builtin_prefetch(ga + 32, 0, 3);
            __builtin_prefetch(gw + 32, 0, 3);
        }
        __syncthreads();                   // previous compute done with LDS
        *reinterpret_cast<uint4*>(&lA[row * 40 + koff])     = a0;
        *reinterpret_cast<uint4*>(&lA[row * 40 + koff + 8]) = a1;
        *reinterpret_cast<uint4*>(&lB[row * 40 + koff])     = b0;
        *reinterpret_cast<uint4*>(&lB[row * 40 + koff + 8]) = b1;
        __syncthreads();

        // A fragments (ISA 16-bit 16x32 layout): lane(m=l16,hi) holds
        //   halfs 0-7 -> K = 8*hi..8*hi+7 ; halfs 8-15 -> K = 16+8*hi..+7
        v16bf af[2];
#pragma unroll
        for (int mt = 0; mt < 2; ++mt) {
            const int base = (wm * 32 + mt * 16 + l16) * 40;
            af[mt] = make_frag(&lA[base + 8 * hi], &lA[base + 16 + 8 * hi]);
        }
        // B fragments (32x16): lane(n=l16,hi) holds halfs 0-7 -> K = 8*hi..8*hi+7 ; halfs 8-15 -> K = 16+8*hi..+7
        v16bf bf[4];
#pragma unroll
        for (int nt = 0; nt < 4; ++nt) {
            const int base = (wn * 64 + nt * 16 + l16) * 40 + 8 * hi;
            bf[nt] = make_frag(&lB[base], &lB[base + 16]);
        }
#pragma unroll
        for (int mt = 0; mt < 2; ++mt)
#pragma unroll
            for (int nt = 0; nt < 4; ++nt)
                acc[mt][nt] = __builtin_amdgcn_wmma_f32_16x16x32_bf16(
                    false, af[mt], false, bf[nt], (short)0, acc[mt][nt],
                    false, false);
    }

    // Epilogue. C/D layout: VGPR r, lanes 0-15 -> M=r ; lanes 16-31 -> M=r+8; N=l16
#pragma unroll
    for (int nt = 0; nt < 4; ++nt) {
        const int n  = bn + wn * 64 + nt * 16 + l16;
        const float bv = bias[n];
#pragma unroll
        for (int mt = 0; mt < 2; ++mt) {
#pragma unroll
            for (int r = 0; r < 8; ++r) {
                const int m = bm + wm * 32 + mt * 16 + r + 8 * hi;
                float v = acc[mt][nt][r] + bv;
                const size_t idx = (size_t)m * N + n;
                if (GELU_OUT) outb[idx] = f2bf(gelu_exact(v));
                else          outf[idx] = v + resid[idx];
            }
        }
    }
}

// ---------------------------------------------------------------- host glue
extern "C" void kernel_launch(void* const* d_in, const int* in_sizes, int n_in,
                              void* d_out, int out_size, void* d_ws, size_t ws_size,
                              hipStream_t stream) {
    (void)in_sizes; (void)n_in; (void)out_size; (void)ws_size;
    const float* x       = (const float*)d_in[0];
    const float* log_tau = (const float*)d_in[1];
    const float* log_dt  = (const float*)d_in[2];
    const float* mix     = (const float*)d_in[3];
    const float* D       = (const float*)d_in[4];
    const float* n1w     = (const float*)d_in[5];
    const float* n1b     = (const float*)d_in[6];
    const float* n2w     = (const float*)d_in[7];
    const float* n2b     = (const float*)d_in[8];
    const float* fc1_w   = (const float*)d_in[9];
    const float* fc1_b   = (const float*)d_in[10];
    const float* fc2_w   = (const float*)d_in[11];
    const float* fc2_b   = (const float*)d_in[12];
    float* out           = (float*)d_out;

    char* ws = (char*)d_ws;
    const size_t off_htr  = 0;                                  //  64 MiB f32 (B,C,T)
    const size_t off_xres = off_htr  + (size_t)BB*CC*TT*4;      //  64 MiB f32 (B,T,C)
    const size_t off_g    = off_xres + (size_t)BB*TT*CC*4;      //  32 MiB bf16 (M,C)
    const size_t off_w1   = off_g    + (size_t)MM*CC*2;         //   8 MiB bf16 (H,C)
    const size_t off_w2   = off_w1   + (size_t)HH*CC*2;         //   8 MiB bf16 (C,H)
    const size_t off_y1   = off_w2   + (size_t)CC*HH*2;         // 128 MiB bf16 (M,H)
    float*          htr  = (float*)(ws + off_htr);
    float*          xres = (float*)(ws + off_xres);
    unsigned short* g    = (unsigned short*)(ws + off_g);
    unsigned short* w1   = (unsigned short*)(ws + off_w1);
    unsigned short* w2   = (unsigned short*)(ws + off_w2);
    unsigned short* y1   = (unsigned short*)(ws + off_y1);
    float*          hrm  = (float*)(ws + off_y1);               // alias: dead before y1 lives

    // 0) weights -> bf16
    k_cvt_bf16<<<2048, 256, 0, stream>>>(fc1_w, w1, HH * CC);
    k_cvt_bf16<<<2048, 256, 0, stream>>>(fc2_w, w2, CC * HH);

    // 1) LN1 -> hrm (B,T,C) f32
    k_layernorm<<<MM, 256, 0, stream>>>(x, n1w, n1b, hrm, nullptr);

    // 2) transpose to (B,C,T) for the scan
    k_transpose_add<<<dim3(CC / 32, TT / 32, BB), dim3(32, 8), 0, stream>>>(
        hrm, nullptr, htr, TT, CC);

    // 3) S4D diagonal recurrence, in place on htr (adds D*u)
    k_s4d_scan<<<BB * CC, 32, 0, stream>>>(htr, log_tau, log_dt, mix, D);

    // 4) transpose back + residual: xres[b,t,c] = x + s4d_out
    k_transpose_add<<<dim3(TT / 32, CC / 32, BB), dim3(32, 8), 0, stream>>>(
        htr, x, xres, CC, TT);

    // 5) LN2 -> g (bf16, M x C)
    k_layernorm<<<MM, 256, 0, stream>>>(xres, n2w, n2b, nullptr, g);

    // 6) GEMM1 + bias + exact GELU -> y1 bf16 (M x H)
    k_gemm_wmma<true><<<dim3(HH / 128, MM / 128), 256, 0, stream>>>(
        g, w1, fc1_b, nullptr, y1, nullptr, CC, HH);

    // 7) GEMM2 + bias + residual -> out f32 (M x C)
    k_gemm_wmma<false><<<dim3(CC / 128, MM / 128), 256, 0, stream>>>(
        y1, w2, fc2_b, xres, nullptr, out, HH, CC);
}
